// PointNet2Classification_85126251807130
// MI455X (gfx1250) — matrix operand order repaired and checked by reading, compiled, not run on hardware
//
#include <hip/hip_runtime.h>
#include <hip/hip_bf16.h>
#include <stdint.h>

// ---------------------------------------------------------------------------
// PointNet++ (SSG) forward for MI455X / gfx1250, wave32 + WMMA f16->f32.
//
// Pipeline:
//   fps(16384->128) -> ball_query/group -> fused MLP[128,128,128]+maxpool
//   fps(128->32)    -> ball_query/group -> fused MLP[128,128,256]+maxpool
//   global group    -> fused MLP[256,256,512]+maxpool
//   cls/sem heads (fp32)
// ---------------------------------------------------------------------------

typedef __attribute__((ext_vector_type(16))) _Float16 hvec16;
typedef __attribute__((ext_vector_type(8)))  _Float16 hvec8;
typedef __attribute__((ext_vector_type(8)))  float    fvec8;

#define NWAVES 8   // 256-thread blocks for the MLP kernels

// ------------------------- WMMA fragment helpers ---------------------------
// A (16x32 f16, MxK): lane ln -> row m0+(ln&15); halves k0+{0|8}..+7 and +16..+23.
__device__ __forceinline__ hvec16 load_a_lds(const _Float16* A, int lda, int m0, int k0) {
    int lane = threadIdx.x & 31;
    const _Float16* p = A + (size_t)(m0 + (lane & 15)) * lda + (k0 + ((lane >> 4) << 3));
    hvec8 lo = *(const hvec8*)p;        // K = kb .. kb+7
    hvec8 hi = *(const hvec8*)(p + 16); // K = kb+16 .. kb+23
    hvec16 r;
#pragma unroll
    for (int i = 0; i < 8; ++i) { r[i] = lo[i]; r[i + 8] = hi[i]; }
    return r;
}

// B (32x16 f16, KxN): weights stored row-major [N][Kpad] (== (out,in) layout).
// lane ln -> col n0+(ln&15); halves k0+{0|8}..+7 and +16..+23.
__device__ __forceinline__ hvec16 load_b_glb(const _Float16* __restrict__ W, int kpad,
                                             int n0, int k0) {
    int lane = threadIdx.x & 31;
    const _Float16* p = W + (size_t)(n0 + (lane & 15)) * kpad + (k0 + ((lane >> 4) << 3));
    hvec8 lo = *(const hvec8*)p;
    hvec8 hi = *(const hvec8*)(p + 16);
    hvec16 r;
#pragma unroll
    for (int i = 0; i < 8; ++i) { r[i] = lo[i]; r[i + 8] = hi[i]; }
    return r;
}

// D (16x16 f32): VGPR r -> row m0 + (lane<16?0:8) + r, col n0 + (lane&15).
// Apply bias + ReLU, store f16 into LDS.
__device__ __forceinline__ void store_d_relu(fvec8 acc, _Float16* Y, int ldy,
                                             int m0, int n0,
                                             const float* __restrict__ bias) {
    int lane = threadIdx.x & 31;
    int n = n0 + (lane & 15);
    int mb = m0 + ((lane >> 4) << 3);
    float bv = bias[n];
#pragma unroll
    for (int r = 0; r < 8; ++r) {
        float v = acc[r] + bv;
        v = v > 0.f ? v : 0.f;
        Y[(size_t)(mb + r) * ldy + n] = (_Float16)v;
    }
}

// One MLP layer: Y[M,N] = relu(A[M,K] * W[N,K]^T + bias). 8 waves split tiles.
template <int M, int K, int N>
__device__ __forceinline__ void gemm_layer(const _Float16* A, int lda,
                                           const _Float16* __restrict__ W,
                                           const float* __restrict__ bias,
                                           _Float16* Y, int ldy) {
    constexpr int TM = M / 16, TN = N / 16, NT = TM * TN;
    int wave = threadIdx.x >> 5;
    for (int t = wave; t < NT; t += NWAVES) {
        int m0 = (t % TM) << 4;
        int n0 = (t / TM) << 4;
        fvec8 acc = {};
#pragma unroll
        for (int k0 = 0; k0 < K; k0 += 32) {
            hvec16 a = load_a_lds(A, lda, m0, k0);
            hvec16 b = load_b_glb(W, K, n0, k0);
            acc = __builtin_amdgcn_wmma_f32_16x16x32_f16(false, a, false, b,
                                                         (short)0, acc, false, false);
        }
        store_d_relu(acc, Y, ldy, m0, n0, bias);
    }
}

// Fused 3-layer shared MLP + maxpool over the M (nsample) axis, one group/block.
template <int M, int K0, int C1, int C2, int C3, bool OUTF32>
__global__ __launch_bounds__(256) void mlp_kernel(
    const _Float16* __restrict__ grouped,  // [grid][M][K0]
    const _Float16* __restrict__ w1, const float* __restrict__ b1,
    const _Float16* __restrict__ w2, const float* __restrict__ b2,
    const _Float16* __restrict__ w3, const float* __restrict__ b3,
    _Float16* __restrict__ feat16, float* __restrict__ feat32) {
    constexpr int LA0 = K0 + 8, LA2 = C2 + 8;   // bufA strides (+8 halves: bank skew)
    constexpr int LB1 = C1 + 8, LB3 = C3 + 8;   // bufB strides
    constexpr int ASZ = M * (LA0 > LA2 ? LA0 : LA2);
    constexpr int BSZ = M * (LB1 > LB3 ? LB1 : LB3);
    __shared__ __align__(16) _Float16 bufA[ASZ];
    __shared__ __align__(16) _Float16 bufB[BSZ];

    int g = blockIdx.x;
    __builtin_prefetch(w1, 0, 1);
    __builtin_prefetch(w2, 0, 1);

    const _Float16* Xg = grouped + (size_t)g * M * K0;
    for (int i = threadIdx.x; i < M * K0; i += 256) {
        int m = i / K0, c = i % K0;
        bufA[m * LA0 + c] = Xg[i];
    }
    __syncthreads();
    gemm_layer<M, K0, C1>(bufA, LA0, w1, b1, bufB, LB1);
    __syncthreads();
    gemm_layer<M, C1, C2>(bufB, LB1, w2, b2, bufA, LA2);
    __syncthreads();
    gemm_layer<M, C2, C3>(bufA, LA2, w3, b3, bufB, LB3);
    __syncthreads();
    for (int c = threadIdx.x; c < C3; c += 256) {
        float mx = -3.4e38f;
        for (int m = 0; m < M; ++m) {
            float v = (float)bufB[m * LB3 + c];
            mx = v > mx ? v : mx;
        }
        if constexpr (OUTF32) feat32[(size_t)g * C3 + c] = mx;
        else                  feat16[(size_t)g * C3 + c] = (_Float16)mx;
    }
}

// ------------------------- Farthest point sampling -------------------------
// One block per batch; points held in registers (PPT per thread, strided).
// Reference semantics: selected[0]=0; selected[i]=argmax(min-dists after i upd).
template <int THREADS, int PPT>
__global__ void fps_kernel(const float* __restrict__ xyz, int n, int stride,
                           int npoint, float* __restrict__ new_xyz) {
    int b = blockIdx.x;
    const float* base = xyz + (size_t)b * n * stride;
    float px[PPT], py[PPT], pz[PPT], pd[PPT];
    int tid = threadIdx.x;
#pragma unroll
    for (int j = 0; j < PPT; ++j) {
        int p = tid + j * THREADS;
        if (p < n) {
            const float* q = base + (size_t)p * stride;
            px[j] = q[0]; py[j] = q[1]; pz[j] = q[2]; pd[j] = 1e10f;
        } else { px[j] = 0.f; py[j] = 0.f; pz[j] = 0.f; pd[j] = -1.f; }
    }
    __shared__ float scx, scy, scz;
    __shared__ float rv[THREADS / 32];
    __shared__ int   ri[THREADS / 32];
    __shared__ int   scur;

    int cur = 0;
    for (int it = 0; it < npoint; ++it) {
        if (tid == (cur % THREADS)) {
            int j = cur / THREADS;
            scx = px[j]; scy = py[j]; scz = pz[j];
        }
        __syncthreads();
        float cx = scx, cy = scy, cz = scz;
        if (tid == 0) {
            float* o = new_xyz + ((size_t)b * npoint + it) * 3;
            o[0] = cx; o[1] = cy; o[2] = cz;
        }
        float bv = -2.f; int bi = 0;
#pragma unroll
        for (int j = 0; j < PPT; ++j) {
            int p = tid + j * THREADS;
            float dx = px[j] - cx, dy = py[j] - cy, dz = pz[j] - cz;
            float d = dx * dx + dy * dy + dz * dz;
            pd[j] = pd[j] < d ? pd[j] : d;
            if (pd[j] > bv || (pd[j] == bv && p < bi)) { bv = pd[j]; bi = p; }
        }
#pragma unroll
        for (int off = 16; off > 0; off >>= 1) {
            float ov = __shfl_down(bv, off);
            int   oi = __shfl_down(bi, off);
            if (ov > bv || (ov == bv && oi < bi)) { bv = ov; bi = oi; }
        }
        int lane = tid & 31, wv = tid >> 5;
        if (lane == 0) { rv[wv] = bv; ri[wv] = bi; }
        __syncthreads();
        if (tid == 0) {
            float best = rv[0]; int bidx = ri[0];
            for (int w = 1; w < THREADS / 32; ++w)
                if (rv[w] > best || (rv[w] == best && ri[w] < bidx)) { best = rv[w]; bidx = ri[w]; }
            scur = bidx;
        }
        __syncthreads();
        cur = scur;
    }
}

// ------------------------- Ball query + grouping ---------------------------
// wave32 ordered scan: first `nsample` in-radius indices in ascending order.
__global__ void group0_kernel(const float* __restrict__ pc,      // [32][16384][6]
                              const float* __restrict__ nxyz,    // [32][128][3]
                              _Float16* __restrict__ grouped) {  // [4096][64][32]
    int g = blockIdx.x, b = g >> 7;
    const float* xyzb = pc + (size_t)b * 16384 * 6;
    float cx = nxyz[g * 3], cy = nxyz[g * 3 + 1], cz = nxyz[g * 3 + 2];
    __shared__ int sel[64];
    __shared__ int scnt;
    int tid = threadIdx.x;
    if (tid == 0) scnt = 0;
    __syncthreads();
    if (tid < 32) {
        int lane = tid, count = 0;
        for (int chunk = 0; chunk < 16384 && count < 64; chunk += 32) {
            int p = chunk + lane;
            const float* q = xyzb + (size_t)p * 6;
            float dx = q[0] - cx, dy = q[1] - cy, dz = q[2] - cz;
            bool in = (dx * dx + dy * dy + dz * dz) < 0.04f;
            unsigned mask = (unsigned)__ballot(in);
            int pos = count + __popc(mask & ((1u << lane) - 1u));
            if (in && pos < 64) sel[pos] = p;
            count += __popc(mask);
        }
        if (lane == 0) scnt = count < 64 ? count : 64;
    }
    __syncthreads();
    int cnt = scnt;
    for (int j = tid; j < 64; j += 64) {
        int idx = (cnt == 0) ? 0 : sel[j < cnt ? j : 0];
        const float* q = xyzb + (size_t)idx * 6;
        _Float16* o = grouped + ((size_t)g * 64 + j) * 32;
        o[0] = (_Float16)(q[0] - cx); o[1] = (_Float16)(q[1] - cy); o[2] = (_Float16)(q[2] - cz);
        o[3] = (_Float16)q[3]; o[4] = (_Float16)q[4]; o[5] = (_Float16)q[5];
        for (int c = 6; c < 32; ++c) o[c] = (_Float16)0.f;
    }
}

__global__ void group1_kernel(const float* __restrict__ nxyz0,   // [32][128][3]
                              const _Float16* __restrict__ f0,   // [32*128][128]
                              const float* __restrict__ nxyz1,   // [32][32][3]
                              _Float16* __restrict__ grouped) {  // [1024][64][160]
    int g = blockIdx.x, b = g >> 5;
    const float* xyzb = nxyz0 + (size_t)b * 128 * 3;
    float cx = nxyz1[g * 3], cy = nxyz1[g * 3 + 1], cz = nxyz1[g * 3 + 2];
    __shared__ int sel[64];
    __shared__ int scnt;
    int tid = threadIdx.x;
    if (tid == 0) scnt = 0;
    __syncthreads();
    if (tid < 32) {
        int lane = tid, count = 0;
        for (int chunk = 0; chunk < 128 && count < 64; chunk += 32) {
            int p = chunk + lane;
            const float* q = xyzb + (size_t)p * 3;
            float dx = q[0] - cx, dy = q[1] - cy, dz = q[2] - cz;
            bool in = (dx * dx + dy * dy + dz * dz) < 0.16f;
            unsigned mask = (unsigned)__ballot(in);
            int pos = count + __popc(mask & ((1u << lane) - 1u));
            if (in && pos < 64) sel[pos] = p;
            count += __popc(mask);
        }
        if (lane == 0) scnt = count < 64 ? count : 64;
    }
    __syncthreads();
    int cnt = scnt;
    for (int j = tid; j < 64; j += 64) {
        int idx = (cnt == 0) ? 0 : sel[j < cnt ? j : 0];
        const float* q = xyzb + (size_t)idx * 3;
        const _Float16* f = f0 + ((size_t)b * 128 + idx) * 128;
        _Float16* o = grouped + ((size_t)g * 64 + j) * 160;
        o[0] = (_Float16)(q[0] - cx); o[1] = (_Float16)(q[1] - cy); o[2] = (_Float16)(q[2] - cz);
        for (int c = 0; c < 128; ++c) o[3 + c] = f[c];
        for (int c = 131; c < 160; ++c) o[c] = (_Float16)0.f;
    }
}

__global__ void group2_kernel(const float* __restrict__ nxyz1,    // [32][32][3]
                              const _Float16* __restrict__ f1,    // [32*32][256]
                              _Float16* __restrict__ grouped) {   // [32][32][288]
    int b = blockIdx.x;
    for (int i = threadIdx.x; i < 32 * 288; i += 256) {
        int p = i / 288, c = i % 288;
        _Float16 v;
        if (c < 3)        v = (_Float16)nxyz1[((size_t)b * 32 + p) * 3 + c];
        else if (c < 259) v = f1[((size_t)b * 32 + p) * 256 + (c - 3)];
        else              v = (_Float16)0.f;
        grouped[(size_t)b * 32 * 288 + i] = v;
    }
}

// ------------------------- fp32 -> padded f16 weights ----------------------
__global__ void convert_w(const float* __restrict__ w, _Float16* __restrict__ out,
                          int O, int Cin, int Kpad) {
    int i = blockIdx.x * blockDim.x + threadIdx.x;
    if (i >= O * Kpad) return;
    int o = i / Kpad, c = i % Kpad;
    out[i] = (c < Cin) ? (_Float16)w[o * Cin + c] : (_Float16)0.f;
}

// ------------------------- Classification heads ----------------------------
__global__ void heads_kernel(const float* __restrict__ feats,  // [32][512]
    const float* cw0, const float* cb0, const float* cw1, const float* cb1,
    const float* cw2, const float* cb2,
    const float* sw0, const float* sb0, const float* sw1, const float* sb1,
    const float* sw2, const float* sb2,
    float* __restrict__ out) {                                 // [32] obj + [32][8] sem
    int b = blockIdx.x, tid = threadIdx.x;
    __shared__ float f[512], h1[256], h2[256];
    for (int i = tid; i < 512; i += 256) f[i] = feats[(size_t)b * 512 + i];
    __syncthreads();
    // cls head
    { float a = cb0[tid]; for (int k = 0; k < 512; ++k) a += cw0[(size_t)tid * 512 + k] * f[k];
      h1[tid] = a > 0.f ? a : 0.f; }
    __syncthreads();
    { float a = cb1[tid]; for (int k = 0; k < 256; ++k) a += cw1[(size_t)tid * 256 + k] * h1[k];
      h2[tid] = a > 0.f ? a : 0.f; }
    __syncthreads();
    if (tid == 0) {
        float a = cb2[0];
        for (int k = 0; k < 256; ++k) a += cw2[k] * h2[k];
        out[b] = a;
    }
    __syncthreads();
    // sem head (reuse h1/h2)
    { float a = sb0[tid]; for (int k = 0; k < 512; ++k) a += sw0[(size_t)tid * 512 + k] * f[k];
      h1[tid] = a > 0.f ? a : 0.f; }
    __syncthreads();
    { float a = sb1[tid]; for (int k = 0; k < 256; ++k) a += sw1[(size_t)tid * 256 + k] * h1[k];
      h2[tid] = a > 0.f ? a : 0.f; }
    __syncthreads();
    if (tid < 8) {
        float a = sb2[tid];
        for (int k = 0; k < 256; ++k) a += sw2[(size_t)tid * 256 + k] * h2[k];
        out[32 + (size_t)b * 8 + tid] = a;
    }
}

// ---------------------------------------------------------------------------
extern "C" void kernel_launch(void* const* d_in, const int* in_sizes, int n_in,
                              void* d_out, int out_size, void* d_ws, size_t ws_size,
                              hipStream_t stream) {
    (void)in_sizes; (void)n_in; (void)out_size; (void)ws_size;
    const float* pc = (const float*)d_in[0];
    const float* saw[3][3]; const float* sab[3][3];
    int p = 1;
    for (int s = 0; s < 3; ++s)
        for (int l = 0; l < 3; ++l) {
            saw[s][l] = (const float*)d_in[p++];
            sab[s][l] = (const float*)d_in[p++];
        }
    const float *cw[3], *cb[3], *sw[3], *sb[3];
    for (int l = 0; l < 3; ++l) { cw[l] = (const float*)d_in[p++]; cb[l] = (const float*)d_in[p++]; }
    for (int l = 0; l < 3; ++l) { sw[l] = (const float*)d_in[p++]; sb[l] = (const float*)d_in[p++]; }

    uint8_t* ws = (uint8_t*)d_ws;
    size_t off = 0;
    auto alloc = [&](size_t bytes) -> void* {
        off = (off + 255) & ~(size_t)255;
        void* r = ws + off;
        off += bytes;
        return r;
    };
    float*    nx0 = (float*)alloc((size_t)32 * 128 * 3 * 4);
    float*    nx1 = (float*)alloc((size_t)32 * 32 * 3 * 4);
    _Float16* g0  = (_Float16*)alloc((size_t)4096 * 64 * 32 * 2);
    _Float16* f0  = (_Float16*)alloc((size_t)4096 * 128 * 2);
    _Float16* g1  = (_Float16*)alloc((size_t)1024 * 64 * 160 * 2);
    _Float16* f1  = (_Float16*)alloc((size_t)1024 * 256 * 2);
    _Float16* g2  = (_Float16*)alloc((size_t)32 * 32 * 288 * 2);
    float*    f2  = (float*)alloc((size_t)32 * 512 * 4);

    const int O[3][3]  = {{128, 128, 128}, {128, 128, 256}, {256, 256, 512}};
    const int CI[3][3] = {{6, 128, 128}, {131, 128, 128}, {259, 256, 256}};
    const int KP[3][3] = {{32, 128, 128}, {160, 128, 128}, {288, 256, 256}};
    _Float16* w16[3][3];
    for (int s = 0; s < 3; ++s)
        for (int l = 0; l < 3; ++l)
            w16[s][l] = (_Float16*)alloc((size_t)O[s][l] * KP[s][l] * 2);

    // fp32 -> f16 (K-padded) weight conversion
    for (int s = 0; s < 3; ++s)
        for (int l = 0; l < 3; ++l) {
            int total = O[s][l] * KP[s][l];
            convert_w<<<(total + 255) / 256, 256, 0, stream>>>(saw[s][l], w16[s][l],
                                                               O[s][l], CI[s][l], KP[s][l]);
        }

    // Stage 0
    fps_kernel<1024, 16><<<32, 1024, 0, stream>>>(pc, 16384, 6, 128, nx0);
    group0_kernel<<<4096, 64, 0, stream>>>(pc, nx0, g0);
    mlp_kernel<64, 32, 128, 128, 128, false><<<4096, 256, 0, stream>>>(
        g0, w16[0][0], sab[0][0], w16[0][1], sab[0][1], w16[0][2], sab[0][2], f0, nullptr);

    // Stage 1
    fps_kernel<128, 1><<<32, 128, 0, stream>>>(nx0, 128, 3, 32, nx1);
    group1_kernel<<<1024, 64, 0, stream>>>(nx0, f0, nx1, g1);
    mlp_kernel<64, 160, 128, 128, 256, false><<<1024, 256, 0, stream>>>(
        g1, w16[1][0], sab[1][0], w16[1][1], sab[1][1], w16[1][2], sab[1][2], f1, nullptr);

    // Stage 2 (global)
    group2_kernel<<<32, 256, 0, stream>>>(nx1, f1, g2);
    mlp_kernel<32, 288, 256, 256, 512, true><<<32, 256, 0, stream>>>(
        g2, w16[2][0], sab[2][0], w16[2][1], sab[2][1], w16[2][2], sab[2][2], nullptr, f2);

    // Heads
    heads_kernel<<<32, 256, 0, stream>>>(f2, cw[0], cb[0], cw[1], cb[1], cw[2], cb[2],
                                         sw[0], sb[0], sw[1], sb[1], sw[2], sb[2],
                                         (float*)d_out);
}
